// CausalSelfAttention_77738908058237
// MI455X (gfx1250) — matrix operand order repaired and checked by reading, not compiled, not run on hardware
//
#include <hip/hip_runtime.h>
#include <hip/hip_bf16.h>
#include <math.h>

// ---------- problem constants ----------
#define BATCH 2
#define SEQ   2048
#define DIM   2048
#define NHEAD 16
#define NKV   4
#define HD    128
#define M_TOT (BATCH*SEQ)          // 4096
#define NQKV  (DIM + 2*NKV*HD)     // 3072
#define SOFTCAP 30.0f
#define INV_SQRT_HD 0.08838834764831845f   // 1/sqrt(128)

typedef __bf16 bf16;
typedef bf16  v16bf __attribute__((ext_vector_type(16)));
typedef float v8f   __attribute__((ext_vector_type(8)));

union Frag16 { v16bf v; uint4 q[2]; };

static __device__ __forceinline__ v8f wmma_bf16f32(v16bf a, v16bf b, v8f c) {
  return __builtin_amdgcn_wmma_f32_16x16x32_bf16(false, a, false, b, (short)0, c, false, false);
}

// A-fragment: lane holds row m, contiguous K chunk hi*16..+16
static __device__ __forceinline__ void load_a(Frag16& f, const bf16* row, int k0, int hi) {
  f.q[0] = *(const uint4*)(row + k0 + hi * 16);
  f.q[1] = *(const uint4*)(row + k0 + hi * 16 + 8);
}
// B-fragment: lane holds col n (row of B^T), contiguous K chunk hi*16..+16
static __device__ __forceinline__ void load_b(Frag16& f, const bf16* bp) {
  f.q[0] = *(const uint4*)(bp);
  f.q[1] = *(const uint4*)(bp + 8);
}

// ---------- kernel 1: fp32 -> bf16 ----------
__global__ void f32_to_bf16_kernel(const float* __restrict__ src, bf16* __restrict__ dst, size_t n) {
  size_t i = (size_t)blockIdx.x * blockDim.x + threadIdx.x;
  if (i < n) dst[i] = (bf16)src[i];
}

// ---------- kernel 2/5: C(MxN) = A(MxK) * B(NxK)^T, bf16 in, fp32 out ----------
// block: 8 waves (2x4), block tile 64x256, wave tile 32x64 (2x4 C fragments),
// K pipelined in steps of 64 with two register buffer sets.
__global__ __launch_bounds__(256) void gemm_bf16_nt(const bf16* __restrict__ A,
                                                    const bf16* __restrict__ B,
                                                    float* __restrict__ C,
                                                    int M, int N, int K) {
  const int lane   = threadIdx.x & 31;
  const int wave   = threadIdx.x >> 5;
  const int lane16 = lane & 15;
  const int hi     = lane >> 4;
  const int m0 = blockIdx.x * 64 + (wave >> 2) * 32;
  const int n0 = blockIdx.y * 256 + (wave & 3) * 64;

  v8f acc[2][4];
#pragma unroll
  for (int i = 0; i < 2; ++i)
#pragma unroll
    for (int j = 0; j < 4; ++j) acc[i][j] = (v8f){0.f,0.f,0.f,0.f,0.f,0.f,0.f,0.f};

  const bf16* arow0 = A + (size_t)(m0 + lane16) * K;
  const bf16* arow1 = A + (size_t)(m0 + 16 + lane16) * K;
  const bf16* brow  = B + (size_t)(n0 + lane16) * K;   // + j*16*K per B fragment

  Frag16 aC[2], bC[4], aN[2], bN[4];
  load_a(aC[0], arow0, 0, hi);
  load_a(aC[1], arow1, 0, hi);
#pragma unroll
  for (int j = 0; j < 4; ++j) load_b(bC[j], brow + (size_t)j * 16 * K + hi * 16);

  for (int k0 = 0; k0 < K; k0 += 64) {
    const int k1 = k0 + 32;
    // prefetch k0+32 into the N buffers while doing math on C buffers
    load_a(aN[0], arow0, k1, hi);
    load_a(aN[1], arow1, k1, hi);
#pragma unroll
    for (int j = 0; j < 4; ++j) load_b(bN[j], brow + (size_t)j * 16 * K + k1 + hi * 16);
#pragma unroll
    for (int j = 0; j < 4; ++j) {
      acc[0][j] = wmma_bf16f32(aC[0].v, bC[j].v, acc[0][j]);
      acc[1][j] = wmma_bf16f32(aC[1].v, bC[j].v, acc[1][j]);
    }
    // prefetch k0+64 into the C buffers while doing math on N buffers
    if (k0 + 64 < K) {
      load_a(aC[0], arow0, k0 + 64, hi);
      load_a(aC[1], arow1, k0 + 64, hi);
#pragma unroll
      for (int j = 0; j < 4; ++j) load_b(bC[j], brow + (size_t)j * 16 * K + k0 + 64 + hi * 16);
    }
#pragma unroll
    for (int j = 0; j < 4; ++j) {
      acc[0][j] = wmma_bf16f32(aN[0].v, bN[j].v, acc[0][j]);
      acc[1][j] = wmma_bf16f32(aN[1].v, bN[j].v, acc[1][j]);
    }
  }

#pragma unroll
  for (int i = 0; i < 2; ++i)
#pragma unroll
    for (int j = 0; j < 4; ++j)
#pragma unroll
      for (int r = 0; r < 8; ++r)
        C[(size_t)(m0 + i * 16 + r + 8 * hi) * N + (n0 + j * 16 + lane16)] = acc[i][j][r];
}

// ---------- kernel 3a: per-(token, head) RMS norm + RoPE + gain, emit bf16 ----------
__global__ __launch_bounds__(32) void qk_post_kernel(const float* __restrict__ qkv,
                                                     const float* __restrict__ qgain,
                                                     bf16* __restrict__ q_bf,
                                                     bf16* __restrict__ k_bf) {
  const int m    = blockIdx.x;
  const int slot = blockIdx.y;
  const int lane = threadIdx.x;
  const int b    = m >> 11;
  const int s    = m & (SEQ - 1);
  const bool isQ = slot < 16;
  const int h    = isQ ? slot : slot - 16;

  const float* base = qkv + (size_t)m * NQKV + (isQ ? 0 : DIM) + h * HD;
  float v[4];
  const float4 xl = *(const float4*)(base + lane * 4);
  v[0] = xl.x; v[1] = xl.y; v[2] = xl.z; v[3] = xl.w;

  // RMS norm over 128
  float ss = v[0]*v[0] + v[1]*v[1] + v[2]*v[2] + v[3]*v[3];
#pragma unroll
  for (int off = 1; off < 32; off <<= 1) ss += __shfl_xor(ss, off, 32);
  const float rs = rsqrtf(ss * (1.0f / HD) + 1.1920929e-7f);
#pragma unroll
  for (int j = 0; j < 4; ++j) v[j] *= rs;

  // RoPE on dims 0..31: pair (d, d+16) -> partner lane = lane ^ 4
  float other[4];
#pragma unroll
  for (int j = 0; j < 4; ++j) other[j] = __shfl_xor(v[j], 4, 32);
  if (lane < 8) {
    const float pos = (float)s;
#pragma unroll
    for (int j = 0; j < 4; ++j) {
      const int d = lane * 4 + j;
      const int i = d & 15;
      const float freq = pos * __expf(-0.5756462732485115f * (float)i); // 10000^(-i/16)
      float sn, cs;
      __sincosf(freq, &sn, &cs);
      const float sign = (d < 16) ? 1.f : -1.f;
      v[j] = v[j] * cs + sign * other[j] * sn;
    }
  }

  if (isQ) {
    const float g = qgain[h] * INV_SQRT_HD;   // fold softmax scale + gain into Q
    bf16* dst = q_bf + ((size_t)m * NHEAD + h) * HD + lane * 4;
#pragma unroll
    for (int j = 0; j < 4; ++j) dst[j] = (bf16)(v[j] * g);
  } else {
    bf16* dst = k_bf + (((size_t)b * NKV + h) * SEQ + s) * HD + lane * 4;
#pragma unroll
    for (int j = 0; j < 4; ++j) dst[j] = (bf16)v[j];
  }
}

// ---------- kernel 3b: V -> bf16, transposed Vt[b][kvh][dim][seq] ----------
__global__ void v_transpose_kernel(const float* __restrict__ qkv, bf16* __restrict__ vT) {
  const size_t i = (size_t)blockIdx.x * blockDim.x + threadIdx.x;
  if (i >= (size_t)M_TOT * NKV * HD) return;
  const int m = (int)(i >> 9);
  const int r = (int)(i & 511);
  const int h = r >> 7;
  const int d = r & 127;
  const int b = m >> 11;
  const int s = m & (SEQ - 1);
  const float val = qkv[(size_t)m * NQKV + DIM + NKV * HD + h * HD + d];
  vT[(((size_t)b * NKV + h) * HD + d) * SEQ + s] = (bf16)val;
}

// ---------- kernel 4: flash attention, 1 wave per 16-row Q tile ----------
__global__ __launch_bounds__(32) void attn_kernel(const bf16* __restrict__ Q,
                                                  const bf16* __restrict__ Km,
                                                  const bf16* __restrict__ Vt,
                                                  bf16* __restrict__ Y) {
  const int qb   = blockIdx.x;          // 16-row q tile
  const int h    = blockIdx.y;          // q head
  const int b    = blockIdx.z;          // batch
  const int kvh  = h >> 2;
  const int lane = threadIdx.x;
  const int lane16 = lane & 15;
  const int hi     = lane >> 4;

  __shared__ __align__(16) bf16 pstage[16 * 32];

  // Q fragments for 16 rows x 128 dims (4 x K=32)
  const bf16* qbase = Q + ((size_t)(b * SEQ + qb * 16 + lane16) * NHEAD + h) * HD;
  Frag16 qf[4];
#pragma unroll
  for (int kk = 0; kk < 4; ++kk) load_a(qf[kk], qbase, kk * 32, hi);

  const bf16* Kbase = Km + ((size_t)b * NKV + kvh) * SEQ * HD;
  const bf16* Vbase = Vt + ((size_t)b * NKV + kvh) * HD * SEQ;

  float m_r[8], l_r[8];
  v8f o[8];
#pragma unroll
  for (int r = 0; r < 8; ++r) { m_r[r] = -INFINITY; l_r[r] = 0.f; }
#pragma unroll
  for (int f = 0; f < 8; ++f) o[f] = (v8f){0.f,0.f,0.f,0.f,0.f,0.f,0.f,0.f};

  const int nblocks = (qb * 16 + 16 + 31) >> 5;   // 32-key blocks covering causal span
  for (int kb = 0; kb < nblocks; ++kb) {
    const int j0 = kb * 32;
    v8f st[2];
#pragma unroll
    for (int t = 0; t < 2; ++t) {
      const int jt = j0 + t * 16;
      // load all 4 K fragments up front, then run WMMAs on 2 independent accumulators
      Frag16 kf[4];
#pragma unroll
      for (int kk = 0; kk < 4; ++kk)
        load_b(kf[kk], Kbase + (size_t)(jt + lane16) * HD + kk * 32 + hi * 16);
      v8f s0 = (v8f){0.f,0.f,0.f,0.f,0.f,0.f,0.f,0.f};
      v8f s1 = (v8f){0.f,0.f,0.f,0.f,0.f,0.f,0.f,0.f};
      s0 = wmma_bf16f32(qf[0].v, kf[0].v, s0);
      s1 = wmma_bf16f32(qf[1].v, kf[1].v, s1);
      s0 = wmma_bf16f32(qf[2].v, kf[2].v, s0);
      s1 = wmma_bf16f32(qf[3].v, kf[3].v, s1);
      v8f sacc = s0 + s1;
      const int key = jt + lane16;
#pragma unroll
      for (int r = 0; r < 8; ++r) {
        const int qg = qb * 16 + r + 8 * hi;
        float val = SOFTCAP * tanhf(sacc[r] * (1.0f / SOFTCAP));
        sacc[r] = (key <= qg) ? val : -INFINITY;
      }
      st[t] = sacc;
    }

    // online softmax row statistics (row spread over 16 lanes of a half-wave)
    float alpha[8], mnew[8];
#pragma unroll
    for (int r = 0; r < 8; ++r) {
      float mx = fmaxf(st[0][r], st[1][r]);
#pragma unroll
      for (int off = 1; off < 16; off <<= 1) mx = fmaxf(mx, __shfl_xor(mx, off, 32));
      mnew[r]  = fmaxf(m_r[r], mx);
      alpha[r] = __expf(m_r[r] - mnew[r]);
      m_r[r]   = mnew[r];
    }
#pragma unroll
    for (int r = 0; r < 8; ++r) {
      const float p0 = __expf(st[0][r] - mnew[r]);
      const float p1 = __expf(st[1][r] - mnew[r]);
      pstage[(r + 8 * hi) * 32 + lane16]      = (bf16)p0;
      pstage[(r + 8 * hi) * 32 + 16 + lane16] = (bf16)p1;
      float ps = p0 + p1;
#pragma unroll
      for (int off = 1; off < 16; off <<= 1) ps += __shfl_xor(ps, off, 32);
      l_r[r] = l_r[r] * alpha[r] + ps;
    }

    asm volatile("s_wait_dscnt 0x0" ::: "memory");

    // reload P as an A-fragment (16x32)
    Frag16 pf;
    const bf16* pp = &pstage[lane16 * 32 + hi * 16];
    pf.q[0] = *(const uint4*)(pp);
    pf.q[1] = *(const uint4*)(pp + 8);

    // load all 8 V fragments, rescale O, accumulate P @ V (8 independent chains)
    Frag16 vf[8];
#pragma unroll
    for (int f = 0; f < 8; ++f)
      load_b(vf[f], Vbase + (size_t)(f * 16 + lane16) * SEQ + j0 + hi * 16);
#pragma unroll
    for (int f = 0; f < 8; ++f) {
      v8f oo = o[f];
#pragma unroll
      for (int r = 0; r < 8; ++r) oo[r] *= alpha[r];
      o[f] = wmma_bf16f32(pf.v, vf[f].v, oo);
    }
  }

  // normalize and emit bf16 y[b][s][h][d]
#pragma unroll
  for (int f = 0; f < 8; ++f) {
#pragma unroll
    for (int r = 0; r < 8; ++r) {
      const float val = o[f][r] / l_r[r];
      Y[((size_t)(b * SEQ + qb * 16 + r + 8 * hi) * NHEAD + h) * HD + f * 16 + lane16] = (bf16)val;
    }
  }
}

// ---------- host ----------
extern "C" void kernel_launch(void* const* d_in, const int* in_sizes, int n_in,
                              void* d_out, int out_size, void* d_ws, size_t ws_size,
                              hipStream_t stream) {
  (void)in_sizes; (void)n_in; (void)out_size; (void)ws_size;
  const float* x     = (const float*)d_in[0];
  const float* Wq    = (const float*)d_in[1];
  const float* Wk    = (const float*)d_in[2];
  const float* Wv    = (const float*)d_in[3];
  const float* Wproj = (const float*)d_in[4];
  const float* qgain = (const float*)d_in[5];

  char* ws = (char*)d_ws;
  size_t off = 0;
  bf16* x_bf  = (bf16*)(ws + off); off += (size_t)M_TOT * DIM * 2;       // 16 MB
  bf16* wqkv  = (bf16*)(ws + off); off += (size_t)NQKV * DIM * 2;        // 12 MB
  bf16* wproj = (bf16*)(ws + off); off += (size_t)DIM * DIM * 2;         // 8 MB
  float* qkv  = (float*)(ws + off); off += (size_t)M_TOT * NQKV * 4;     // 50 MB
  bf16* q_bf  = (bf16*)(ws + off); off += (size_t)M_TOT * DIM * 2;       // 16 MB
  bf16* k_bf  = (bf16*)(ws + off); off += (size_t)BATCH * NKV * SEQ * HD * 2; // 4 MB
  bf16* vT    = (bf16*)(ws + off); off += (size_t)BATCH * NKV * HD * SEQ * 2; // 4 MB
  bf16* y_bf  = (bf16*)(ws + off); off += (size_t)M_TOT * DIM * 2;       // 16 MB

  // 1) conversions (Wq/Wk/Wv concatenated row-wise -> single 3072x2048 weight)
  {
    size_t n;
    n = (size_t)M_TOT * DIM;
    f32_to_bf16_kernel<<<(unsigned)((n + 255) / 256), 256, 0, stream>>>(x, x_bf, n);
    n = (size_t)DIM * DIM;
    f32_to_bf16_kernel<<<(unsigned)((n + 255) / 256), 256, 0, stream>>>(Wq, wqkv, n);
    n = (size_t)NKV * HD * DIM;
    f32_to_bf16_kernel<<<(unsigned)((n + 255) / 256), 256, 0, stream>>>(Wk, wqkv + (size_t)DIM * DIM, n);
    f32_to_bf16_kernel<<<(unsigned)((n + 255) / 256), 256, 0, stream>>>(Wv, wqkv + (size_t)DIM * DIM + (size_t)NKV * HD * DIM, n);
    n = (size_t)DIM * DIM;
    f32_to_bf16_kernel<<<(unsigned)((n + 255) / 256), 256, 0, stream>>>(Wproj, wproj, n);
  }

  // 2) fused QKV projection: (4096 x 2048) @ (3072 x 2048)^T -> fp32
  gemm_bf16_nt<<<dim3(M_TOT / 64, NQKV / 256), 256, 0, stream>>>(x_bf, wqkv, qkv, M_TOT, NQKV, DIM);

  // 3) RMS norm + RoPE + gain (Q,K) and V transpose, all to bf16
  qk_post_kernel<<<dim3(M_TOT, NHEAD + NKV), 32, 0, stream>>>(qkv, qgain, q_bf, k_bf);
  {
    size_t n = (size_t)M_TOT * NKV * HD;
    v_transpose_kernel<<<(unsigned)((n + 255) / 256), 256, 0, stream>>>(qkv, vT);
  }

  // 4) causal flash attention with softcap
  attn_kernel<<<dim3(SEQ / 16, NHEAD, BATCH), 32, 0, stream>>>(q_bf, k_bf, vT, y_bf);

  // 5) output projection: (4096 x 2048) @ (2048 x 2048)^T -> fp32 d_out
  gemm_bf16_nt<<<dim3(M_TOT / 64, DIM / 256), 256, 0, stream>>>(y_bf, wproj, (float*)d_out, M_TOT, DIM, DIM);
}
